// Transformer_block_69114613727775
// MI455X (gfx1250) — matrix operand order repaired and checked by reading, not compiled, not run on hardware
//
#include <hip/hip_runtime.h>

// ---------------------------------------------------------------------------
// MI455X (gfx1250) transformer block.
// Strategy: cast weights/activations to f16, run all GEMMs + attention on
// v_wmma_f32_16x16x32_f16 (f32 accumulate). Problem working set (~50MB f16)
// fits in the 192MB L2, so register-tiled WMMA with direct global loads is the
// right data path; HBM roofline is ~5us for the whole block at 23.3 TB/s.
// Round 3: TDM builtin uses this toolchain's 6-arg signature
// (g0 u32x4, g1 i32x8, g2 i32x4, g3 i32x4, i32x8, cpol).
// ---------------------------------------------------------------------------

typedef __attribute__((ext_vector_type(16))) _Float16 v16h;
typedef __attribute__((ext_vector_type(8)))  _Float16 v8h;
typedef __attribute__((ext_vector_type(8)))  float    v8f;
typedef unsigned int u32x4 __attribute__((ext_vector_type(4)));
typedef int          i32x8 __attribute__((ext_vector_type(8)));
typedef int          i32x4 __attribute__((ext_vector_type(4)));

#define D_MODEL 1024
#define NHEAD   16
#define DKH     64
#define DFF     4096
#define SEQ     2048
#define BATCH   2
#define MROWS   (BATCH * SEQ)   // 4096 token rows

#define WMMA_F16(A, B, C) \
  __builtin_amdgcn_wmma_f32_16x16x32_f16(false, (A), false, (B), (short)0, (C), false, false)

// TDM availability (device pass only; host pass falls back automatically).
#if defined(__AMDGCN__) && __has_builtin(__builtin_amdgcn_tensor_load_to_lds) && \
    __has_builtin(__builtin_amdgcn_s_wait_tensorcnt)
#define USE_TDM 1
#else
#define USE_TDM 0
#endif

// ---- WMMA fragment loaders (wave32; A and B both use contiguous k = 16*(l>>4)+j) -----------
// A (16x32 f16): lane l holds row M=(l&15); elements j = K[kb + j],
// kb = (l>>4)*16 (one contiguous 32B run).
__device__ __forceinline__ v16h load_a_frag(const _Float16* base, int ld,
                                            int row, int k0, int lane) {
  const _Float16* r = base + (size_t)row * ld + k0 + ((lane >> 4) << 4);
  union { v16h v; v8h h[2]; } u;
  u.h[0] = *(const v8h*)(r);
  u.h[1] = *(const v8h*)(r + 8);
  return u.v;
}

// B (32x16 f16), sourced from row-major W[N][K]: lane l holds column
// n=(l&15); elements j = K[k0 + (l>>4)*16 + j] (one contiguous 32B run).
__device__ __forceinline__ v16h load_b_frag(const _Float16* base, int ld,
                                            int nrow, int k0, int lane) {
  const _Float16* r = base + (size_t)nrow * ld + k0 + ((lane >> 4) << 4);
  return *(const v16h*)(r);
}

// ---- elementwise helpers ---------------------------------------------------
__global__ void cast_f32_f16_kernel(const float* __restrict__ s,
                                    _Float16* __restrict__ d, int n) {
  int i = blockIdx.x * 256 + threadIdx.x;
  if (i < n) d[i] = (_Float16)s[i];
}

__global__ void add_f32_kernel(const float* __restrict__ a,
                               const float* __restrict__ b,
                               float* __restrict__ o, int n) {
  int i = blockIdx.x * 256 + threadIdx.x;
  if (i < n) o[i] = a[i] + b[i];
}

// RMSNorm over D=1024, one block (256 thr) per token row, f16 output.
// The 4KB row is staged once into LDS by the Tensor Data Mover (1D tile,
// data_size=4B, tile_dim0=tensor_dim0=1024), then both passes read LDS.
__global__ void rmsnorm_cast_kernel(const float* __restrict__ x,
                                    const float* __restrict__ w,
                                    _Float16* __restrict__ out) {
  extern __shared__ __align__(16) float xrow[];   // 1024 floats, LDS base = 0
  int row = blockIdx.x;
  const float* xr = x + (size_t)row * D_MODEL;

#if USE_TDM
  if (threadIdx.x < 32) {                          // one wave issues the DMA
    unsigned long long ga = (unsigned long long)(const void*)xr;
    // D# group0: count=1 | lds_addr=0 | global_addr | type=2 ("image")
    u32x4 g0 = { 1u,
                 0u,
                 (unsigned)(ga & 0xFFFFFFFFu),
                 (unsigned)((ga >> 32) & 0x01FFFFFFu) | (2u << 30) };
    // D# group1: data_size=4B; tensor_dim0=1024; tensor_dim1=1;
    //            tile_dim0=1024; tensor_dim0_stride=1024.
    i32x8 g1 = { (int)(2u << 16),
                 (int)((D_MODEL & 0xFFFFu) << 16),
                 (int)((D_MODEL >> 16) | (1u << 16)),
                 (int)((unsigned)D_MODEL << 16),
                 0,
                 D_MODEL,
                 0,
                 0 };
    i32x4 z4 = { 0, 0, 0, 0 };
    i32x8 z8 = { 0, 0, 0, 0, 0, 0, 0, 0 };
    __builtin_amdgcn_tensor_load_to_lds(g0, g1, z4, z4, z8, 0);
    __builtin_amdgcn_s_wait_tensorcnt(0);
  }
  __syncthreads();
#else
  for (int i = threadIdx.x; i < D_MODEL; i += 256) xrow[i] = xr[i];
  __syncthreads();
#endif

  float s = 0.f;
  for (int i = threadIdx.x; i < D_MODEL; i += 256) { float v = xrow[i]; s += v * v; }
#pragma unroll
  for (int off = 16; off > 0; off >>= 1) s += __shfl_xor(s, off, 32);
  __shared__ float red[8];
  if ((threadIdx.x & 31) == 0) red[threadIdx.x >> 5] = s;
  __syncthreads();
  float tot = 0.f;
#pragma unroll
  for (int i = 0; i < 8; ++i) tot += red[i];
  float inv = rsqrtf(tot * (1.0f / D_MODEL) + 1e-5f);
  _Float16* orow = out + (size_t)row * D_MODEL;
  for (int i = threadIdx.x; i < D_MODEL; i += 256)
    orow[i] = (_Float16)(xrow[i] * inv * w[i]);
}

// RoPE + pack [M,1024] f32 -> [b*h][s][dk] f16. One thread per even/odd pair.
__global__ void rope_pack_kernel(const float* __restrict__ src,
                                 _Float16* __restrict__ dst) {
  int idx = blockIdx.x * 256 + threadIdx.x;          // MROWS * 512 pairs
  if (idx >= MROWS * (D_MODEL / 2)) return;
  int row = idx >> 9;                                 // token row
  int p   = idx & 511;
  int h   = p >> 5;                                   // head
  int i   = p & 31;                                   // freq index (DK/2=32)
  int b   = row >> 11;
  int s   = row & (SEQ - 1);
  float freq = __powf(10000.f, -(float)i / 32.f);
  float ang  = (float)s * freq;
  float c = __cosf(ang), sn = __sinf(ang);
  const float* sp = src + (size_t)row * D_MODEL + h * DKH + 2 * i;
  float xe = sp[0], xo = sp[1];
  _Float16* dp = dst + (((size_t)(b * NHEAD + h) * SEQ + s) * DKH) + 2 * i;
  dp[0] = (_Float16)(c * xe - sn * xo);
  dp[1] = (_Float16)(sn * xe + c * xo);
}

// Pack V transposed: [M,1024] f32 -> [b*h][dk][s] f16 (so PV matmul reads
// contiguous-K B fragments).
__global__ void vpack_t_kernel(const float* __restrict__ src,
                               _Float16* __restrict__ dst) {
  int idx = blockIdx.x * 256 + threadIdx.x;          // MROWS * D_MODEL
  if (idx >= MROWS * D_MODEL) return;
  int row = idx >> 10;
  int e   = idx & (D_MODEL - 1);
  int b   = row >> 11;
  int s   = row & (SEQ - 1);
  int h   = e >> 6;
  int dk  = e & (DKH - 1);
  dst[(((size_t)(b * NHEAD + h) * DKH + dk) * SEQ) + s] = (_Float16)src[idx];
}

// ---- generic WMMA GEMM: C[M,N] (f32) = A[M,K] (f16) * W[N,K]^T (f16) -------
// Block = 4 waves (2x2), block tile 128x128, wave tile 64x64, K step 32.
// global_prefetch_b8 pulls the following cache line (k0+64) per fragment row.
__global__ __launch_bounds__(128) void gemm_nt_kernel(
    const _Float16* __restrict__ A, const _Float16* __restrict__ W,
    float* __restrict__ C, int M, int N, int K) {
  const int lane = threadIdx.x & 31;
  const int wave = threadIdx.x >> 5;
  const int nl   = lane & 15;
  const int m0   = blockIdx.y * 128 + (wave >> 1) * 64;
  const int n0   = blockIdx.x * 128 + (wave & 1) * 64;
  (void)M;
  v8f acc[4][4] = {};
  for (int k0 = 0; k0 < K; k0 += 32) {
    v16h a[4], b[4];
#pragma unroll
    for (int i = 0; i < 4; ++i) {
      const _Float16* ra = A + (size_t)(m0 + i * 16 + nl) * K + k0;
      a[i] = load_a_frag(A, K, m0 + i * 16 + nl, k0, lane);
      __builtin_prefetch(ra + 64, 0, 1);   // next cache line of this row
    }
#pragma unroll
    for (int j = 0; j < 4; ++j) {
      const _Float16* rb = W + (size_t)(n0 + j * 16 + nl) * K + k0;
      b[j] = load_b_frag(W, K, n0 + j * 16 + nl, k0, lane);
      __builtin_prefetch(rb + 64, 0, 1);
    }
#pragma unroll
    for (int i = 0; i < 4; ++i)
#pragma unroll
      for (int j = 0; j < 4; ++j) acc[i][j] = WMMA_F16(a[i], b[j], acc[i][j]);
  }
  const int mb = (lane >> 4) << 3;
#pragma unroll
  for (int i = 0; i < 4; ++i)
#pragma unroll
    for (int j = 0; j < 4; ++j) {
      int mrow = m0 + i * 16 + mb;
      int ncol = n0 + j * 16 + nl;
#pragma unroll
      for (int e = 0; e < 8; ++e) C[(size_t)(mrow + e) * N + ncol] = acc[i][j][e];
    }
}

// ---- fused SwiGLU GEMM: act16[M,DFF] = silu(h2*W1^T) * (h2*W3^T) -----------
// Wave tile 32x64 (two accumulator sets), block = 4 waves stacked in M.
__global__ __launch_bounds__(128) void ffn_swiglu_kernel(
    const _Float16* __restrict__ A, const _Float16* __restrict__ W1,
    const _Float16* __restrict__ W3, _Float16* __restrict__ OutAct) {
  const int lane = threadIdx.x & 31;
  const int wave = threadIdx.x >> 5;
  const int nl   = lane & 15;
  const int m0   = blockIdx.y * 128 + wave * 32;
  const int n0   = blockIdx.x * 64;
  v8f au[2][4] = {}, ag[2][4] = {};
  for (int k0 = 0; k0 < D_MODEL; k0 += 32) {
    v16h a[2], b1[4], b3[4];
#pragma unroll
    for (int i = 0; i < 2; ++i) {
      a[i] = load_a_frag(A, D_MODEL, m0 + i * 16 + nl, k0, lane);
      __builtin_prefetch(A + (size_t)(m0 + i * 16 + nl) * D_MODEL + k0 + 64, 0, 1);
    }
#pragma unroll
    for (int j = 0; j < 4; ++j) {
      b1[j] = load_b_frag(W1, D_MODEL, n0 + j * 16 + nl, k0, lane);
      b3[j] = load_b_frag(W3, D_MODEL, n0 + j * 16 + nl, k0, lane);
      __builtin_prefetch(W1 + (size_t)(n0 + j * 16 + nl) * D_MODEL + k0 + 64, 0, 1);
      __builtin_prefetch(W3 + (size_t)(n0 + j * 16 + nl) * D_MODEL + k0 + 64, 0, 1);
    }
#pragma unroll
    for (int i = 0; i < 2; ++i)
#pragma unroll
      for (int j = 0; j < 4; ++j) {
        au[i][j] = WMMA_F16(a[i], b1[j], au[i][j]);
        ag[i][j] = WMMA_F16(a[i], b3[j], ag[i][j]);
      }
  }
  const int mb = (lane >> 4) << 3;
#pragma unroll
  for (int i = 0; i < 2; ++i)
#pragma unroll
    for (int j = 0; j < 4; ++j) {
      int mrow = m0 + i * 16 + mb;
      int ncol = n0 + j * 16 + nl;
#pragma unroll
      for (int e = 0; e < 8; ++e) {
        float u = au[i][j][e];
        float g = ag[i][j][e];
        float act = (u / (1.f + __expf(-u))) * g;   // silu(u) * g
        OutAct[(size_t)(mrow + e) * DFF + ncol] = (_Float16)act;
      }
    }
}

// ---- causal flash attention ------------------------------------------------
// Grid (S/64, B*H), block 128 = 4 waves; wave owns 16 q rows, online softmax,
// 32-key chunks. Score WMMAs: Q(16x32d) x K^T; PV WMMAs: P(16x32s) x V^T.
// P tile is moved C-layout -> A-layout through a per-wave LDS staging buffer.
__device__ __forceinline__ float redmax16(float v) {
#pragma unroll
  for (int off = 1; off < 16; off <<= 1) v = fmaxf(v, __shfl_xor(v, off, 32));
  return v;
}
__device__ __forceinline__ float redsum16(float v) {
#pragma unroll
  for (int off = 1; off < 16; off <<= 1) v += __shfl_xor(v, off, 32);
  return v;
}

__global__ __launch_bounds__(128) void attn_kernel(
    const _Float16* __restrict__ q, const _Float16* __restrict__ k,
    const _Float16* __restrict__ vt, _Float16* __restrict__ o16) {
  const int lane = threadIdx.x & 31;
  const int wave = threadIdx.x >> 5;
  const int nl   = lane & 15;
  const int hh   = lane >> 4;
  const int qt   = blockIdx.x;
  const int bh   = blockIdx.y;
  const int b    = bh >> 4;
  const int h    = bh & (NHEAD - 1);
  const int qbase = qt * 64 + wave * 16;
  const int rbase = qbase + (hh << 3);            // this half-wave's 8 q rows

  const _Float16* qp = q  + (size_t)bh * SEQ * DKH;
  const _Float16* kp = k  + (size_t)bh * SEQ * DKH;
  const _Float16* vp = vt + (size_t)bh * DKH * SEQ;

  __shared__ __align__(16) _Float16 plds[4][16 * 32];
  _Float16* my = plds[wave];

  v16h qa[2];
  qa[0] = load_a_frag(qp, DKH, qbase + nl, 0, lane);
  qa[1] = load_a_frag(qp, DKH, qbase + nl, 32, lane);

  v8f oacc[4] = {};
  float mrow[8], lrow[8];
#pragma unroll
  for (int i = 0; i < 8; ++i) { mrow[i] = -1e30f; lrow[i] = 0.f; }

  for (int kc = 0; kc * 32 <= qbase + 15; ++kc) {
    const int key0 = kc * 32;
    v8f s0 = {}, s1 = {};
#pragma unroll
    for (int d = 0; d < 2; ++d) {
      v16h kb0 = load_b_frag(kp, DKH, key0 + nl,      d * 32, lane);
      v16h kb1 = load_b_frag(kp, DKH, key0 + 16 + nl, d * 32, lane);
      s0 = WMMA_F16(qa[d], kb0, s0);
      s1 = WMMA_F16(qa[d], kb1, s1);
    }
    const int col0 = key0 + nl;
    const int col1 = col0 + 16;
    float csv[8], p0[8], p1[8];
#pragma unroll
    for (int i = 0; i < 8; ++i) {
      int row = rbase + i;
      float a0 = (col0 <= row) ? s0[i] * 0.125f : -1e30f;  // 1/sqrt(64)
      float a1 = (col1 <= row) ? s1[i] * 0.125f : -1e30f;
      float mx = redmax16(fmaxf(a0, a1));
      float nm = fmaxf(mrow[i], mx);
      float e0 = __expf(a0 - nm), e1 = __expf(a1 - nm);
      float rs = redsum16(e0 + e1);
      float cs = __expf(mrow[i] - nm);
      lrow[i] = lrow[i] * cs + rs;
      mrow[i] = nm;
      csv[i] = cs;
      p0[i] = e0; p1[i] = e1;
    }
#pragma unroll
    for (int j = 0; j < 4; ++j)
#pragma unroll
      for (int i = 0; i < 8; ++i) oacc[j][i] *= csv[i];
    // C-layout -> A-layout via LDS (per-wave private region, no barrier).
#pragma unroll
    for (int i = 0; i < 8; ++i) {
      my[(i + (hh << 3)) * 32 + nl]      = (_Float16)p0[i];
      my[(i + (hh << 3)) * 32 + 16 + nl] = (_Float16)p1[i];
    }
    v16h pa = load_a_frag(my, 32, nl, 0, lane);
#pragma unroll
    for (int j = 0; j < 4; ++j) {
      v16h vb = load_b_frag(vp, SEQ, j * 16 + nl, key0, lane);
      oacc[j] = WMMA_F16(pa, vb, oacc[j]);
    }
  }
  // Normalize and write o packed as [b, s, h*64+dk] f16 (ready for W_o GEMM).
#pragma unroll
  for (int j = 0; j < 4; ++j)
#pragma unroll
    for (int i = 0; i < 8; ++i) {
      int srow = rbase + i;
      float val = oacc[j][i] / lrow[i];
      o16[((size_t)(b * SEQ + srow)) * D_MODEL + h * DKH + j * 16 + nl] =
          (_Float16)val;
    }
}

// ---------------------------------------------------------------------------
extern "C" void kernel_launch(void* const* d_in, const int* in_sizes, int n_in,
                              void* d_out, int out_size, void* d_ws, size_t ws_size,
                              hipStream_t stream) {
  (void)in_sizes; (void)n_in; (void)out_size; (void)ws_size;
  const float* x    = (const float*)d_in[0];
  const float* w_q  = (const float*)d_in[1];
  const float* w_k  = (const float*)d_in[2];
  const float* w_v  = (const float*)d_in[3];
  const float* w_o  = (const float*)d_in[4];
  const float* ln1w = (const float*)d_in[5];
  const float* ln2w = (const float*)d_in[6];
  const float* w1   = (const float*)d_in[7];
  const float* w2   = (const float*)d_in[8];
  const float* w3   = (const float*)d_in[9];
  float* out = (float*)d_out;

  char* ws = (char*)d_ws;
  const size_t MB = 1u << 20;
  _Float16* wq16  = (_Float16*)(ws + 0 * MB);    // 2 MiB each (1024x1024 f16)
  _Float16* wk16  = (_Float16*)(ws + 2 * MB);
  _Float16* wv16  = (_Float16*)(ws + 4 * MB);
  _Float16* wo16  = (_Float16*)(ws + 6 * MB);
  _Float16* w1_16 = (_Float16*)(ws + 8 * MB);    // 8 MiB each (4096x1024 f16)
  _Float16* w3_16 = (_Float16*)(ws + 16 * MB);
  _Float16* w2_16 = (_Float16*)(ws + 24 * MB);
  _Float16* h16   = (_Float16*)(ws + 32 * MB);   // 8 MiB  (4096x1024 f16)
  float*    tmpf  = (float*)   (ws + 40 * MB);   // 16 MiB (4096x1024 f32)
  _Float16* q16   = (_Float16*)(ws + 56 * MB);   // 8 MiB
  _Float16* k16   = (_Float16*)(ws + 64 * MB);   // 8 MiB
  _Float16* vt16  = (_Float16*)(ws + 72 * MB);   // 8 MiB
  _Float16* o16   = (_Float16*)(ws + 80 * MB);   // 8 MiB
  float*    x2f   = (float*)   (ws + 88 * MB);   // 16 MiB
  _Float16* h2_16 = (_Float16*)(ws + 104 * MB);  // 8 MiB
  _Float16* act16 = (_Float16*)(ws + 112 * MB);  // 32 MiB (4096x4096 f16)

  const int nDD  = D_MODEL * D_MODEL;   // 1M
  const int nDF  = D_MODEL * DFF;       // 4M
  const int nTok = MROWS * D_MODEL;     // 4M

  // 1) cast weights to f16
  cast_f32_f16_kernel<<<(nDD + 255) / 256, 256, 0, stream>>>(w_q, wq16, nDD);
  cast_f32_f16_kernel<<<(nDD + 255) / 256, 256, 0, stream>>>(w_k, wk16, nDD);
  cast_f32_f16_kernel<<<(nDD + 255) / 256, 256, 0, stream>>>(w_v, wv16, nDD);
  cast_f32_f16_kernel<<<(nDD + 255) / 256, 256, 0, stream>>>(w_o, wo16, nDD);
  cast_f32_f16_kernel<<<(nDF + 255) / 256, 256, 0, stream>>>(w1, w1_16, nDF);
  cast_f32_f16_kernel<<<(nDF + 255) / 256, 256, 0, stream>>>(w3, w3_16, nDF);
  cast_f32_f16_kernel<<<(nDF + 255) / 256, 256, 0, stream>>>(w2, w2_16, nDF);

  // 2) h = rmsnorm(x)  (TDM-staged row in LDS, 4KB dynamic shared)
  rmsnorm_cast_kernel<<<MROWS, 256, D_MODEL * sizeof(float), stream>>>(x, ln1w, h16);

  dim3 gK(D_MODEL / 128, MROWS / 128);   // (8, 32) for D->D GEMMs
  // 3) Q/K/V projections + RoPE/pack
  gemm_nt_kernel<<<gK, 128, 0, stream>>>(h16, wq16, tmpf, MROWS, D_MODEL, D_MODEL);
  rope_pack_kernel<<<(MROWS * 512 + 255) / 256, 256, 0, stream>>>(tmpf, q16);
  gemm_nt_kernel<<<gK, 128, 0, stream>>>(h16, wk16, tmpf, MROWS, D_MODEL, D_MODEL);
  rope_pack_kernel<<<(MROWS * 512 + 255) / 256, 256, 0, stream>>>(tmpf, k16);
  gemm_nt_kernel<<<gK, 128, 0, stream>>>(h16, wv16, tmpf, MROWS, D_MODEL, D_MODEL);
  vpack_t_kernel<<<(nTok + 255) / 256, 256, 0, stream>>>(tmpf, vt16);

  // 4) causal attention
  attn_kernel<<<dim3(SEQ / 64, BATCH * NHEAD), 128, 0, stream>>>(q16, k16, vt16, o16);

  // 5) output projection + residual
  gemm_nt_kernel<<<gK, 128, 0, stream>>>(o16, wo16, tmpf, MROWS, D_MODEL, D_MODEL);
  add_f32_kernel<<<(nTok + 255) / 256, 256, 0, stream>>>(x, tmpf, x2f, nTok);

  // 6) FFN: rmsnorm -> fused SwiGLU -> down projection -> residual
  rmsnorm_cast_kernel<<<MROWS, 256, D_MODEL * sizeof(float), stream>>>(x2f, ln2w, h2_16);
  ffn_swiglu_kernel<<<dim3(DFF / 64, MROWS / 128), 128, 0, stream>>>(h2_16, w1_16, w3_16, act16);
  gemm_nt_kernel<<<gK, 128, 0, stream>>>(act16, w2_16, tmpf, MROWS, D_MODEL, DFF);
  add_f32_kernel<<<(nTok + 255) / 256, 256, 0, stream>>>(x2f, tmpf, out, nTok);
}
